// MyGraphConvModel_2903397893093
// MI455X (gfx1250) — compile-verified, matrix operand order checked by reading, not run on hardware
//
#include <hip/hip_runtime.h>
#include <hip/hip_bf16.h>

#define NATOMS 300000
#define DMAX   6
#define FDIM   75
#define BMOL   16384
#define C1D    16
#define C2D    32
#define HDIM   256
#define EPSBN  1e-3f

typedef __attribute__((ext_vector_type(16))) _Float16 v16h;
typedef __attribute__((ext_vector_type(8)))  float    v8f;

// leaky_relu(x, 0.2) == max(x, 0.2*x)  (branch-free, no cndmask)
__device__ __forceinline__ float leaky(float x) { return fmaxf(x, 0.2f * x); }

// ---------------------------------------------------------------------------
// Fused GraphConv: gather+mask neighbor sum, per-degree matvec, bias, leaky,
// batchnorm. C threads per atom (c = out channel), 256-thread blocks.
// ---------------------------------------------------------------------------
template<int FIN, int C>
__global__ void gc_kernel(const float* __restrict__ x, const int* __restrict__ adj,
                          const int* __restrict__ deg,
                          const float* __restrict__ Ws, const float* __restrict__ bs,
                          const float* __restrict__ Wn, const float* __restrict__ bnb,
                          const float* __restrict__ g, const float* __restrict__ b,
                          const float* __restrict__ m, const float* __restrict__ v,
                          float* __restrict__ out, int n_atoms)
{
    constexpr int ATOMS = 256 / C;
    __shared__ float xs[ATOMS][FIN];
    __shared__ float ns[ATOMS][FIN];
    const int tid  = threadIdx.x;
    const int base = blockIdx.x * ATOMS;

    // Phase A: cooperatively stage self features and masked neighbor sums.
    for (int e = tid; e < ATOMS * FIN; e += 256) {
        const int a = e / FIN, f = e % FIN;
        const int n = base + a;
        float sv = 0.f, nsum = 0.f;
        if (n < n_atoms) {
            sv = x[(size_t)n * FIN + f];
            const int dg = deg[n];
            #pragma unroll
            for (int d = 0; d < DMAX; ++d)
                if (d < dg) nsum += x[(size_t)adj[n * DMAX + d] * FIN + f];
        }
        xs[a][f] = sv;
        ns[a][f] = nsum;
    }
    __syncthreads();

    // Phase B: per-degree matvec + bias + leaky + BN.
    const int c = tid % C, a = tid / C;
    const int n = base + a;
    if (n < n_atoms) {
        const int dg = deg[n];
        const float* ws = Ws + (size_t)dg * FIN * C + c;
        const float* wn = Wn + (size_t)dg * FIN * C + c;
        float acc = bs[dg * C + c] + bnb[dg * C + c];
        for (int f = 0; f < FIN; ++f)
            acc += xs[a][f] * ws[(size_t)f * C] + ns[a][f] * wn[(size_t)f * C];
        acc = leaky(acc);
        const float sc = g[c] * rsqrtf(v[c] + EPSBN);
        out[(size_t)n * C + c] = (acc - m[c]) * sc + b[c];
    }
}

// ---------------------------------------------------------------------------
// GraphPool: elementwise max over {self} U valid neighbors.
// ---------------------------------------------------------------------------
template<int C>
__global__ void pool_kernel(const float* __restrict__ x, const int* __restrict__ adj,
                            const int* __restrict__ deg, float* __restrict__ out, int n_atoms)
{
    const long long i = (long long)blockIdx.x * blockDim.x + threadIdx.x;
    if (i >= (long long)n_atoms * C) return;
    const int n = (int)(i / C), c = (int)(i % C);
    float mx = x[i];
    const int dg = deg[n];
    #pragma unroll
    for (int d = 0; d < DMAX; ++d)
        if (d < dg) mx = fmaxf(mx, x[(size_t)adj[n * DMAX + d] * C + c]);
    out[i] = mx;
}

// ---------------------------------------------------------------------------
// Dense 32->256 via v_wmma_f32_16x16x32_f16, fused bias + leaky + BN3.
// 8 waves/block; each wave owns one 16-atom M tile; K=32 = one WMMA deep.
// Weights are staged into LDS pre-swizzled into exact B-fragment order so the
// hot loop reads each fragment as one contiguous 32B LDS vector load.
// ---------------------------------------------------------------------------
__global__ void dense_wmma_kernel(const float* __restrict__ x, const float* __restrict__ W,
                                  const float* __restrict__ bias,
                                  const float* __restrict__ g, const float* __restrict__ b,
                                  const float* __restrict__ m, const float* __restrict__ v,
                                  float* __restrict__ out, int n_atoms)
{
    // B fragments in fragment order: [nt][lane][h] -> 16*32*16 halves = 16 KB
    __shared__ __align__(32) _Float16 lds_w[16 * 32 * 16];
    __shared__ _Float16 lds_a[8][16 * C2D];  // per-wave 16 atoms x 32 feats
    __shared__ float lds_bb[HDIM], lds_sc[HDIM], lds_sh[HDIM];

    const int tid = threadIdx.x;

    // Stage weights pre-swizzled: dst (nt, lane, h) <- W[k(h,lane)][nt*16 + (lane&15)]
    for (int i = tid; i < 16 * 32 * 16; i += 256) {
        const int nt = i >> 9, ln = (i >> 4) & 31, h = i & 15;
        const int hi = h >> 1, lo = h & 1;
        const int k  = ((hi & 3) << 1) + ((hi >> 2) << 4) + (((ln >> 4) & 1) << 3) + lo;
        const int n  = nt * 16 + (ln & 15);
        lds_w[i] = (_Float16)W[k * HDIM + n];
    }
    // Per-channel epilogue constants (blockDim == HDIM == 256).
    {
        const float sc = g[tid] * rsqrtf(v[tid] + EPSBN);
        lds_bb[tid] = bias[tid];
        lds_sc[tid] = sc;
        lds_sh[tid] = b[tid] - m[tid] * sc;
    }

    const int wave   = tid >> 5;
    const int lane   = tid & 31;
    const int base_m = (blockIdx.x * 8 + wave) * 16;

    // Stage A tile (coalesced), guarded for tail waves.
    for (int i = lane; i < 16 * C2D; i += 32) {
        const int gm = base_m + i / C2D;
        lds_a[wave][i] = (gm < n_atoms) ? (_Float16)x[(size_t)gm * C2D + i % C2D]
                                        : (_Float16)0.f;
    }
    __syncthreads();

    // Tail waves exit uniformly (EXEC stays all-1s for live waves' WMMAs).
    // NATOMS % 16 == 0, so every surviving tile is full: no store guards.
    if (base_m >= n_atoms) return;

    // Build A fragment per the 16-bit 16x32 layout.
    const int mloc = lane & 15;
    const int kb   = ((lane >> 4) & 1) << 3;
    v16h afrag;
    #pragma unroll
    for (int h = 0; h < 16; ++h) {
        const int hi = h >> 1, lo = h & 1;
        const int k  = ((hi & 3) << 1) + ((hi >> 2) << 4) + kb + lo;
        afrag[h] = lds_a[wave][mloc * C2D + k];
    }

    const int   nbase = lane & 15;
    const int   row0  = base_m + (((lane >> 4) & 1) << 3);
    float* const op0  = out + (size_t)row0 * HDIM + nbase;

    #pragma unroll
    for (int nt = 0; nt < 16; ++nt) {
        const v16h bfrag = *(const v16h*)(lds_w + ((nt * 32 + lane) << 4));
        v8f cfrag = {};
        cfrag = __builtin_amdgcn_wmma_f32_16x16x32_f16(
            /*neg_a=*/false, afrag, /*neg_b=*/false, bfrag,
            /*c_mod=*/(short)0, cfrag, /*reuse_a=*/false, /*reuse_b=*/false);

        const int   nn = nt * 16 + nbase;
        const float bb = lds_bb[nn];
        const float sc = lds_sc[nn];
        const float sh = lds_sh[nn];
        float* const op = op0 + nt * 16;
        #pragma unroll
        for (int r = 0; r < 8; ++r) {
            const float t = cfrag[r] + bb;
            op[(size_t)r * HDIM] = fmaxf(t, 0.2f * t) * sc + sh;
        }
    }
}

// ---------------------------------------------------------------------------
// GraphGather + leaky + regression: one block per molecule, 256 threads = channels.
// ---------------------------------------------------------------------------
__global__ void gather_kernel(const float* __restrict__ h, const int* __restrict__ mem,
                              const float* __restrict__ regW, const float* __restrict__ regb,
                              float* __restrict__ out, int n_atoms)
{
    __shared__ float red[256];
    const int bm = blockIdx.x;
    const int c  = threadIdx.x;

    int lo = 0, hi = n_atoms;
    while (lo < hi) { int mid = (lo + hi) >> 1; if (mem[mid] < bm) lo = mid + 1; else hi = mid; }
    const int start = lo;
    hi = n_atoms;
    while (lo < hi) { int mid = (lo + hi) >> 1; if (mem[mid] < bm + 1) lo = mid + 1; else hi = mid; }
    const int end = lo;

    float s = 0.f, mx = -__builtin_inff();
    for (int n = start; n < end; ++n) {
        const float val = h[(size_t)n * HDIM + c];
        s += val;
        mx = fmaxf(mx, val);
    }
    if (end == start) mx = 0.f;  // empty-molecule guard (matches reference)

    red[c] = leaky(s) * regW[c] + leaky(mx) * regW[HDIM + c];
    __syncthreads();
    for (int st = 128; st > 0; st >>= 1) {
        if (c < st) red[c] += red[c + st];
        __syncthreads();
    }
    if (c == 0) out[bm] = red[0] + regb[0];
}

// ---------------------------------------------------------------------------
extern "C" void kernel_launch(void* const* d_in, const int* in_sizes, int n_in,
                              void* d_out, int out_size, void* d_ws, size_t ws_size,
                              hipStream_t stream)
{
    const float* af   = (const float*)d_in[0];
    const int*   adj  = (const int*)  d_in[1];
    const int*   deg  = (const int*)  d_in[2];
    const int*   mem  = (const int*)  d_in[3];
    const float* g1Ws = (const float*)d_in[4];
    const float* g1bs = (const float*)d_in[5];
    const float* g1Wn = (const float*)d_in[6];
    const float* g1bn = (const float*)d_in[7];
    const float* g2Ws = (const float*)d_in[8];
    const float* g2bs = (const float*)d_in[9];
    const float* g2Wn = (const float*)d_in[10];
    const float* g2bn = (const float*)d_in[11];
    const float* bn1g = (const float*)d_in[12];
    const float* bn1b = (const float*)d_in[13];
    const float* bn1m = (const float*)d_in[14];
    const float* bn1v = (const float*)d_in[15];
    const float* bn2g = (const float*)d_in[16];
    const float* bn2b = (const float*)d_in[17];
    const float* bn2m = (const float*)d_in[18];
    const float* bn2v = (const float*)d_in[19];
    const float* bn3g = (const float*)d_in[20];
    const float* bn3b = (const float*)d_in[21];
    const float* bn3m = (const float*)d_in[22];
    const float* bn3v = (const float*)d_in[23];
    const float* d1W  = (const float*)d_in[24];
    const float* d1b  = (const float*)d_in[25];
    const float* regW = (const float*)d_in[26];
    const float* regb = (const float*)d_in[27];

    float* ws   = (float*)d_ws;
    float* x1   = ws;                                 // [N, 16]
    float* x1p  = x1  + (size_t)NATOMS * C1D;         // [N, 16]
    float* x2   = x1p + (size_t)NATOMS * C1D;         // [N, 32]
    float* x2p  = x2  + (size_t)NATOMS * C2D;         // [N, 32]
    float* hbuf = x2p + (size_t)NATOMS * C2D;         // [N, 256]
    float* outp = (float*)d_out;                      // [B, 1]

    dim3 blk(256);

    gc_kernel<FDIM, C1D><<<(NATOMS + 15) / 16, blk, 0, stream>>>(
        af, adj, deg, g1Ws, g1bs, g1Wn, g1bn, bn1g, bn1b, bn1m, bn1v, x1, NATOMS);

    pool_kernel<C1D><<<((size_t)NATOMS * C1D + 255) / 256, blk, 0, stream>>>(
        x1, adj, deg, x1p, NATOMS);

    gc_kernel<C1D, C2D><<<(NATOMS + 7) / 8, blk, 0, stream>>>(
        x1p, adj, deg, g2Ws, g2bs, g2Wn, g2bn, bn2g, bn2b, bn2m, bn2v, x2, NATOMS);

    pool_kernel<C2D><<<((size_t)NATOMS * C2D + 255) / 256, blk, 0, stream>>>(
        x2, adj, deg, x2p, NATOMS);

    const int ntiles = (NATOMS + 15) / 16;            // 18750 full tiles
    dense_wmma_kernel<<<(ntiles + 7) / 8, blk, 0, stream>>>(
        x2p, d1W, d1b, bn3g, bn3b, bn3m, bn3v, hbuf, NATOMS);

    gather_kernel<<<BMOL, blk, 0, stream>>>(hbuf, mem, regW, regb, outp, NATOMS);
}
